// RegionProposalNetworkSingleFeature_39109972197420
// MI455X (gfx1250) — matrix operand order repaired and checked by reading, not compiled, not run on hardware
//
#include <hip/hip_runtime.h>

typedef __attribute__((ext_vector_type(16))) _Float16 v16h;
typedef __attribute__((ext_vector_type(8)))  _Float16 v8h;
typedef __attribute__((ext_vector_type(8)))  float    v8f;

constexpr int B_ = 8, C_ = 256, H_ = 128, W_ = 128, A_ = 9;
constexpr int HW_    = H_ * W_;        // 16384
constexpr int NANCH  = HW_ * A_;       // 147456
constexpr int K_     = C_ * 9;         // 2304
constexpr int PRE    = 2000;
constexpr int POST   = 1000;
constexpr int PAD    = 2048;
constexpr float NMS_TH     = 0.7f;
constexpr float MIN_SIZE   = 1.0f;
constexpr float NEG_INF_F  = -1e30f;
constexpr float XFORM_CLIP = 4.135166556742356f;   // log(1000/16)
constexpr int NBLK = NANCH / 256;                  // 576 blocks/image for flat sweeps

#if defined(__gfx1250__) && __has_builtin(__builtin_amdgcn_global_load_async_to_lds_b64)
#define RPN_HAVE_ASYNC_LDS 1
#else
#define RPN_HAVE_ASYNC_LDS 0
#endif

// 64-bit payload type matching the builtin's prototype (gcc vector_size spelling)
typedef int rpn_v2i __attribute__((vector_size(2 * sizeof(int))));

__device__ __forceinline__ void rpn_wait_async0() {
#if __has_builtin(__builtin_amdgcn_s_wait_asynccnt)
  __builtin_amdgcn_s_wait_asynccnt(0);
#else
  asm volatile("s_wait_asynccnt 0" ::: "memory");
#endif
}

// ---------------------------------------------------------------------------
// float -> order-preserving unsigned key (larger float => larger key)
__device__ __forceinline__ unsigned f2key(float f) {
  unsigned u = __float_as_uint(f);
  return u ^ ((u & 0x80000000u) ? 0xFFFFFFFFu : 0x80000000u);
}

// ---------------------------------------------------------------------------
// Pack conv weights f32 (C,C,3,3) -> f16 [co][k], k = (ky*3+kx)*256 + ci
__global__ void k_pack_weights(const float* __restrict__ w, _Float16* __restrict__ wf) {
  int i = blockIdx.x * 256 + threadIdx.x;
  if (i >= C_ * K_) return;
  int co = i / K_, k = i % K_;
  int kk = k / C_, ci = k % C_;
  int ky = kk / 3, kx = kk % 3;
  wf[i] = (_Float16)w[((co * C_ + ci) * 3 + ky) * 3 + kx];
}

// Pack features f32 NCHW -> f16 NHWC
__global__ void k_pack_feat(const float* __restrict__ f, _Float16* __restrict__ ff) {
  size_t i = (size_t)blockIdx.x * 256 + threadIdx.x;   // NHWC flat index
  int ci = (int)(i & 255);
  int x  = (int)((i >> 8)  & 127);
  int y  = (int)((i >> 15) & 127);
  int b  = (int)(i >> 22);
  ff[i] = (_Float16)f[(((size_t)b * C_ + ci) * H_ + y) * W_ + x];
}

// ---------------------------------------------------------------------------
// Fused 3x3 conv (implicit GEMM, WMMA f16->f32) + bias + ReLU + cls/reg heads.
// Block = 256 threads (8 wave32s). Block tile: M=256 out channels, N=32 positions.
// Double-buffered im2col slice in LDS, filled via async global->LDS when available.
__global__ __launch_bounds__(256) void k_conv_rpn(
    const _Float16* __restrict__ wf, const _Float16* __restrict__ ff,
    const float* __restrict__ convb,
    const float* __restrict__ clsw, const float* __restrict__ clsb,
    const float* __restrict__ regw, const float* __restrict__ regb,
    float* __restrict__ obj, float* __restrict__ dlt)
{
  __shared__ __align__(64) _Float16 Blds[2][32][32];  // ping-pong [pos][ci-slice]
  __shared__ float Hlds[256][33];                     // relu(conv) tile, padded stride

  const int tid  = threadIdx.x;
  const int wave = tid >> 5;
  const int lane = tid & 31;
  const int lhi  = lane & 16;   // selects K/M half per ISA fragment layout
  const int lmod = lane & 15;

  const int blk  = blockIdx.x;        // b*512 + tile
  const int b    = blk >> 9;
  const int tile = blk & 511;         // 128 rows x 4 x-tiles
  const int y    = tile >> 2;
  const int x0   = (tile & 3) << 5;

  const int m0 = wave * 32;

  // staging thread mapping: each thread owns one 8-byte slot of the 32x32 tile
  const int p  = tid >> 3;           // position 0..31
  const int c4 = (tid & 7) << 2;     // ci offset 0,4,...,28

  // --- incrementally-carried staging state (next k-step to stage) ----------
  const _Float16* s_ptr = nullptr;
  bool s_ok = false;
  int  s_cs = 0;           // ci-step within tap (0..7)
  int  s_ky = -1, s_kx = -1;

  auto tap_init = [&](int ky, int kx) {
    const int yy = y + ky;
    const int xx = x0 + p + kx;
    s_ok  = (yy >= 0 && yy < H_ && xx >= 0 && xx < W_);
    s_ptr = ff + ((((size_t)b * H_ + yy) * W_ + xx) << 8) + c4;
  };
  auto stage_adv = [&]() {
    if (++s_cs == 8) {
      s_cs = 0;
      if (++s_kx == 2) { s_kx = -1; ++s_ky; }
      tap_init(s_ky, s_kx);
    } else {
      s_ptr += 32;
    }
  };
  auto stage = [&](int bb) {
#if RPN_HAVE_ASYNC_LDS
    if (s_ok) {
      __builtin_amdgcn_global_load_async_to_lds_b64(
          (__attribute__((address_space(1))) rpn_v2i*)(void*)s_ptr,
          (__attribute__((address_space(3))) rpn_v2i*)(void*)&Blds[bb][p][c4],
          0, 0);
    } else {
      *(unsigned long long*)&Blds[bb][p][c4] = 0ull;
    }
#else
    unsigned long long v = 0ull;
    if (s_ok) v = *(const unsigned long long*)s_ptr;
    *(unsigned long long*)&Blds[bb][p][c4] = v;
#endif
  };

  // weight fragment pointers: perfectly linear, +32 halfs per k-step
  const _Float16* wr0 = wf + (size_t)(m0 + lmod) * K_ + (lhi ? 16 : 0);
  const _Float16* wr1 = wr0 + (size_t)16 * K_;

  v8f acc[2][2] = {};   // [m-subtile][n-subtile]
  union U { v16h v; v8h h[2]; };

  // prologue: stage k-step 0 into buffer 0
  tap_init(-1, -1);
  stage(0);
  stage_adv();

  for (int ks = 0; ks < 72; ++ks) {
#if RPN_HAVE_ASYNC_LDS
    rpn_wait_async0();          // own slice of buf[ks&1] landed
#endif
    __syncthreads();            // everyone's slice landed; prev reads of other buf done
    if (ks < 71) {              // overlap: fill the other buffer during compute
      stage((ks + 1) & 1);
      stage_adv();
    }

    U a0, a1, bb0, bb1;
    a0.h[0] = *(const v8h*)wr0;
    a0.h[1] = *(const v8h*)(wr0 + 8);
    a1.h[0] = *(const v8h*)wr1;
    a1.h[1] = *(const v8h*)(wr1 + 8);
    __builtin_prefetch(wr0 + 64, 0, 1);   // next-next k-step weight rows
    __builtin_prefetch(wr1 + 64, 0, 1);
    wr0 += 32;
    wr1 += 32;

    const int bb = ks & 1;
    bb0.v = *(const v16h*)&Blds[bb][lmod][lhi];        // N tile 0
    bb1.v = *(const v16h*)&Blds[bb][16 + lmod][lhi];   // N tile 1

    acc[0][0] = __builtin_amdgcn_wmma_f32_16x16x32_f16(false, a0.v, false, bb0.v, (short)0, acc[0][0], false, false);
    acc[0][1] = __builtin_amdgcn_wmma_f32_16x16x32_f16(false, a0.v, false, bb1.v, (short)0, acc[0][1], false, false);
    acc[1][0] = __builtin_amdgcn_wmma_f32_16x16x32_f16(false, a1.v, false, bb0.v, (short)0, acc[1][0], false, false);
    acc[1][1] = __builtin_amdgcn_wmma_f32_16x16x32_f16(false, a1.v, false, bb1.v, (short)0, acc[1][1], false, false);
  }

  // bias + ReLU into LDS (D layout: lane<16 holds M j, lane>=16 holds M j+8)
  #pragma unroll
  for (int mt = 0; mt < 2; ++mt) {
    const int mbase = m0 + mt * 16 + (lhi ? 8 : 0);
    #pragma unroll
    for (int nt = 0; nt < 2; ++nt) {
      const int n = nt * 16 + lmod;
      #pragma unroll
      for (int j = 0; j < 8; ++j) {
        float v = acc[mt][nt][j] + convb[mbase + j];
        Hlds[mbase + j][n] = v > 0.f ? v : 0.f;
      }
    }
  }
  __syncthreads();

  // heads: 9 cls + 36 reg channels for 32 positions
  for (int oi = tid; oi < 45 * 32; oi += 256) {
    const int o = oi >> 5;
    const int n = oi & 31;
    const float* wrow;
    float s;
    if (o < 9) { wrow = clsw + o * 256;       s = clsb[o]; }
    else       { wrow = regw + (o - 9) * 256; s = regb[o - 9]; }
    for (int c = 0; c < 256; ++c) s += wrow[c] * Hlds[c][n];
    const size_t pos = (size_t)y * W_ + (x0 + n);
    if (o < 9) {
      obj[(size_t)b * NANCH + pos * 9 + o] = s;
    } else {
      const int r = o - 9, a = r >> 2, cd = r & 3;
      dlt[((size_t)b * NANCH + pos * 9 + a) * 4 + cd] = s;
    }
  }
}

// ---------------------------------------------------------------------------
// top-k (2000) via 4-pass MSB radix threshold search
__global__ void k_init(unsigned* hist, unsigned* state, unsigned* cnt,
                       int* selidx, float* selscore) {
  int b = blockIdx.x, t = threadIdx.x;
  if (t == 0) {
    state[b * 4 + 0] = 0;       // prefix
    state[b * 4 + 1] = PRE;     // remaining rank
    cnt[b * 2 + 0] = 0;
    cnt[b * 2 + 1] = 0;
  }
  hist[b * 256 + t] = 0;
  for (int i = t; i < PAD; i += 256) {
    selidx[b * PAD + i] = -1;
    selscore[b * PAD + i] = NEG_INF_F;
  }
}

__global__ void k_hist(const float* __restrict__ obj, unsigned* __restrict__ hist,
                       const unsigned* __restrict__ state, int pass) {
  __shared__ unsigned lh[256];
  const int t = threadIdx.x;
  lh[t] = 0;
  __syncthreads();
  const int blk = blockIdx.x;
  const int b = blk / NBLK;
  const int i = (blk % NBLK) * 256 + t;
  const unsigned key = f2key(obj[(size_t)b * NANCH + i]);
  const unsigned prefix = state[b * 4 + 0];
  const bool ok = (pass == 0) || ((key >> (32 - 8 * pass)) == prefix);
  if (ok) atomicAdd(&lh[(key >> (24 - 8 * pass)) & 255], 1u);
  __syncthreads();
  if (lh[t]) atomicAdd(&hist[b * 256 + t], lh[t]);
}

__global__ void k_scan(unsigned* hist, unsigned* state, int pass) {
  const int b = threadIdx.x;
  if (b >= B_) return;
  unsigned r = state[b * 4 + 1];
  unsigned* h = hist + b * 256;
  unsigned pick = 0;
  for (int v = 255; v >= 0; --v) {
    unsigned c = h[v];
    if (c >= r) { pick = (unsigned)v; break; }
    r -= c;
  }
  state[b * 4 + 0] = (state[b * 4 + 0] << 8) | pick;
  state[b * 4 + 1] = r;
  for (int v = 0; v < 256; ++v) h[v] = 0;
  (void)pass;
}

__global__ void k_select(const float* __restrict__ obj, const unsigned* __restrict__ state,
                         unsigned* cnt, int* selidx, float* selscore) {
  const int blk = blockIdx.x;
  const int b = blk / NBLK;
  const int i = (blk % NBLK) * 256 + threadIdx.x;
  const float s = obj[(size_t)b * NANCH + i];
  const unsigned key = f2key(s);
  const unsigned T = state[b * 4 + 0];
  const unsigned r = state[b * 4 + 1];
  const unsigned gtn = PRE - r;
  if (key > T) {
    unsigned q = atomicAdd(&cnt[b * 2 + 0], 1u);
    if (q < (unsigned)PRE) { selidx[b * PAD + q] = i; selscore[b * PAD + q] = s; }
  } else if (key == T) {
    unsigned e = atomicAdd(&cnt[b * 2 + 1], 1u);
    if (e < r) { selidx[b * PAD + gtn + e] = i; selscore[b * PAD + gtn + e] = s; }
  }
}

// bitonic sort 2048 entries: descending score, ties by ascending index
__global__ __launch_bounds__(1024) void k_sort(int* selidx, float* selscore) {
  __shared__ float ss[PAD];
  __shared__ int   si[PAD];
  const int b = blockIdx.x, t = threadIdx.x;
  for (int i = t; i < PAD; i += 1024) { ss[i] = selscore[b * PAD + i]; si[i] = selidx[b * PAD + i]; }
  __syncthreads();
  for (int k = 2; k <= PAD; k <<= 1) {
    for (int j = k >> 1; j > 0; j >>= 1) {
      for (int i = t; i < PAD; i += 1024) {
        const int q = i ^ j;
        if (q > i) {
          const float sa = ss[i], sb = ss[q];
          const int   ia = si[i], ib = si[q];
          const bool gt = (sa < sb) || (sa == sb && ia > ib);  // "a after b"
          const bool lt = (sa > sb) || (sa == sb && ia < ib);
          const bool doswap = ((i & k) == 0) ? gt : lt;
          if (doswap) { ss[i] = sb; ss[q] = sa; si[i] = ib; si[q] = ia; }
        }
      }
      __syncthreads();
    }
  }
  for (int i = t; i < PAD; i += 1024) { selscore[b * PAD + i] = ss[i]; selidx[b * PAD + i] = si[i]; }
}

// decode + clip + min-size filter
__global__ void k_decode(const int* __restrict__ selidx, const float* __restrict__ selscore,
                         const float* __restrict__ dlt, const int* __restrict__ ih,
                         const int* __restrict__ iw, float* __restrict__ boxes,
                         float* __restrict__ vscore) {
  const int b = blockIdx.x, t = threadIdx.x;
  const float imh = (float)ih[0], imw = (float)iw[0];
  for (int e = t; e < PAD; e += 256) {
    const int idx = selidx[b * PAD + e];
    float s = selscore[b * PAD + e];
    float x1 = 0.f, y1 = 0.f, x2 = 0.f, y2 = 0.f;
    if (idx >= 0) {
      const int a = idx % 9, pos = idx / 9;
      const int xx = pos % W_, yy = pos / W_;
      const float sizes[3] = {64.f, 128.f, 256.f};
      const float rroot[3] = {0.70710678118654752f, 1.f, 1.41421356237309505f};
      const float aw = sizes[a / 3] / rroot[a % 3];
      const float ah = sizes[a / 3] * rroot[a % 3];
      const float ax = xx * (imw / W_);
      const float ay = yy * (imh / H_);
      const float* d = dlt + ((size_t)b * NANCH + idx) * 4;
      const float dx = d[0], dy = d[1];
      const float dw = fminf(d[2], XFORM_CLIP);
      const float dh = fminf(d[3], XFORM_CLIP);
      const float px = dx * aw + ax, py = dy * ah + ay;
      const float pw = __expf(dw) * aw, ph = __expf(dh) * ah;
      x1 = fminf(fmaxf(px - 0.5f * pw, 0.f), imw);
      y1 = fminf(fmaxf(py - 0.5f * ph, 0.f), imh);
      x2 = fminf(fmaxf(px + 0.5f * pw, 0.f), imw);
      y2 = fminf(fmaxf(py + 0.5f * ph, 0.f), imh);
      if (!((x2 - x1 >= MIN_SIZE) && (y2 - y1 >= MIN_SIZE))) s = NEG_INF_F;
    } else {
      s = NEG_INF_F;
    }
    float* bx = boxes + ((size_t)b * PAD + e) * 4;
    bx[0] = x1; bx[1] = y1; bx[2] = x2; bx[3] = y2;
    vscore[b * PAD + e] = s;
  }
}

// greedy NMS over sorted 2000 + top-1000 emission (mimics lax.top_k tie order)
__global__ __launch_bounds__(256) void k_nms(const float* __restrict__ boxes,
                                             const float* __restrict__ vscore,
                                             float* __restrict__ out) {
  __shared__ float bx1[PRE], by1[PRE], bx2[PRE], by2[PRE], sc[PRE], area[PRE];
  __shared__ unsigned char keep[PRE];
  const int b = blockIdx.x, t = threadIdx.x;
  for (int i = t; i < PRE; i += 256) {
    const float* q = boxes + ((size_t)b * PAD + i) * 4;
    bx1[i] = q[0]; by1[i] = q[1]; bx2[i] = q[2]; by2[i] = q[3];
    area[i] = (q[2] - q[0]) * (q[3] - q[1]);
    sc[i]   = vscore[b * PAD + i];
    keep[i] = 1;
  }
  __syncthreads();
  for (int i = 0; i < PRE; ++i) {
    if (keep[i]) {
      const float X1 = bx1[i], Y1 = by1[i], X2 = bx2[i], Y2 = by2[i], AR = area[i];
      for (int j = i + 1 + t; j < PRE; j += 256) {
        if (keep[j]) {
          const float lx = fmaxf(X1, bx1[j]), ly = fmaxf(Y1, by1[j]);
          const float rx = fminf(X2, bx2[j]), ry = fminf(Y2, by2[j]);
          const float w = fmaxf(rx - lx, 0.f), h = fmaxf(ry - ly, 0.f);
          const float inter = w * h;
          const float iou = inter / fmaxf(AR + area[j] - inter, 1e-9f);
          if (iou > NMS_TH) keep[j] = 0;
        }
      }
    }
    __syncthreads();
  }
  if (t == 0) {
    float* props  = out + (size_t)b * POST * 4;
    float* scores = out + (size_t)B_ * POST * 4 + (size_t)b * POST;
    int cnt = 0;
    for (int i = 0; i < PRE && cnt < POST; ++i) {
      if (keep[i] && sc[i] > NEG_INF_F * 0.5f) {
        props[cnt * 4 + 0] = bx1[i]; props[cnt * 4 + 1] = by1[i];
        props[cnt * 4 + 2] = bx2[i]; props[cnt * 4 + 3] = by2[i];
        scores[cnt] = sc[i]; ++cnt;
      }
    }
    for (int i = 0; i < PRE && cnt < POST; ++i) {
      if (!(keep[i] && sc[i] > NEG_INF_F * 0.5f)) {
        props[cnt * 4 + 0] = bx1[i]; props[cnt * 4 + 1] = by1[i];
        props[cnt * 4 + 2] = bx2[i]; props[cnt * 4 + 3] = by2[i];
        scores[cnt] = NEG_INF_F; ++cnt;
      }
    }
    for (; cnt < POST; ++cnt) {
      props[cnt * 4 + 0] = 0.f; props[cnt * 4 + 1] = 0.f;
      props[cnt * 4 + 2] = 0.f; props[cnt * 4 + 3] = 0.f;
      scores[cnt] = NEG_INF_F;
    }
  }
}

// ---------------------------------------------------------------------------
extern "C" void kernel_launch(void* const* d_in, const int* in_sizes, int n_in,
                              void* d_out, int out_size, void* d_ws, size_t ws_size,
                              hipStream_t stream) {
  const float* features = (const float*)d_in[0];
  const float* conv_w   = (const float*)d_in[1];
  const float* conv_b   = (const float*)d_in[2];
  const float* cls_w    = (const float*)d_in[3];
  const float* cls_b    = (const float*)d_in[4];
  const float* reg_w    = (const float*)d_in[5];
  const float* reg_b    = (const float*)d_in[6];
  const int*   image_h  = (const int*)d_in[7];
  const int*   image_w  = (const int*)d_in[8];
  (void)in_sizes; (void)n_in; (void)out_size; (void)ws_size;

  char* ws = (char*)d_ws;
  size_t off = 0;
  auto take = [&](size_t bytes) -> char* {
    char* q = ws + off;
    off += (bytes + 255) & ~(size_t)255;
    return q;
  };

  _Float16* wf16   = (_Float16*)take((size_t)C_ * K_ * sizeof(_Float16));
  _Float16* ff16   = (_Float16*)take((size_t)B_ * H_ * W_ * C_ * sizeof(_Float16));
  float*    obj    = (float*)take((size_t)B_ * NANCH * sizeof(float));
  float*    dlt    = (float*)take((size_t)B_ * NANCH * 4 * sizeof(float));
  unsigned* hist   = (unsigned*)take((size_t)B_ * 256 * sizeof(unsigned));
  unsigned* state  = (unsigned*)take((size_t)B_ * 4 * sizeof(unsigned));
  unsigned* cnt    = (unsigned*)take((size_t)B_ * 2 * sizeof(unsigned));
  int*      selidx = (int*)take((size_t)B_ * PAD * sizeof(int));
  float*    selsc  = (float*)take((size_t)B_ * PAD * sizeof(float));
  float*    boxes  = (float*)take((size_t)B_ * PAD * 4 * sizeof(float));
  float*    vscore = (float*)take((size_t)B_ * PAD * sizeof(float));

  k_pack_weights<<<(C_ * K_ + 255) / 256, 256, 0, stream>>>(conv_w, wf16);
  k_pack_feat<<<(int)(((size_t)B_ * H_ * W_ * C_) / 256), 256, 0, stream>>>(features, ff16);

  k_conv_rpn<<<B_ * 512, 256, 0, stream>>>(wf16, ff16, conv_b, cls_w, cls_b,
                                           reg_w, reg_b, obj, dlt);

  k_init<<<B_, 256, 0, stream>>>(hist, state, cnt, selidx, selsc);
  for (int p = 0; p < 4; ++p) {
    k_hist<<<B_ * NBLK, 256, 0, stream>>>(obj, hist, state, p);
    k_scan<<<1, 32, 0, stream>>>(hist, state, p);
  }
  k_select<<<B_ * NBLK, 256, 0, stream>>>(obj, state, cnt, selidx, selsc);
  k_sort<<<B_, 1024, 0, stream>>>(selidx, selsc);
  k_decode<<<B_, 256, 0, stream>>>(selidx, selsc, dlt, image_h, image_w, boxes, vscore);
  k_nms<<<B_, 256, 0, stream>>>(boxes, vscore, (float*)d_out);
}
